// Seq2SeqLSTM_24790551232944
// MI455X (gfx1250) — matrix operand order repaired and checked by reading, not compiled, not run on hardware
//
#include <hip/hip_runtime.h>
#include <hip/hip_bf16.h>
#include <math.h>

#define B_   2048
#define T_   512
#define F_   21
#define H_   128
#define HOR_ 15
#define NG   512      // 4*H gate width
#define BM   16       // batch rows per workgroup
#define WAVES 4       // waves per block, each owns 32/WAVES gate n-tiles
#define NTW  (32/WAVES)
#define XSTR 40       // padded LDS strides (bank-conflict-free fragment loads)
#define HSTR 136
#define GSTR 516

typedef _Float16 v8h  __attribute__((ext_vector_type(8)));
typedef _Float16 v16h __attribute__((ext_vector_type(16)));
typedef float    v8f  __attribute__((ext_vector_type(8)));

// fast pointwise: v_exp + v_rcp based (1 TRANS + ~3 VALU each)
__device__ __forceinline__ float sigm(float x){
  return __builtin_amdgcn_rcpf(1.0f + __expf(-x));
}
__device__ __forceinline__ float tanh_fast(float x){
  // tanh(x) = 2/(1+exp(-2x)) - 1
  return fmaf(2.0f, __builtin_amdgcn_rcpf(1.0f + __expf(-2.0f*x)), -1.0f);
}

// A-fragment (16x32 f16, ISA 7.12.2): lane = M%16 ; lanes 0-15 hold K {0..7,16..23},
// lanes 16-31 hold K {8..15,24..31}. Two contiguous 16B LDS loads per fragment.
__device__ __forceinline__ v16h load_a_frag(const _Float16* lds, int ld, int kbase, int lane){
  const int row = lane & 15, hi = lane >> 4;
  const _Float16* p = lds + row*ld + kbase + hi*8;
  v8h lo = *(const v8h*)p;
  v8h hb = *(const v8h*)(p + 16);
  v16h a;
  #pragma unroll
  for (int j=0;j<8;++j){ a[j]=lo[j]; a[8+j]=hb[j]; }
  return a;
}

// B-fragment pre-swizzled in workspace: each lane reads its 16 halfs (32B) contiguously.
__device__ __forceinline__ void wmma_acc(v8f& acc, const v16h& a, const _Float16* wt, int lane){
  v16h b = *(const v16h*)(wt + lane*16);
  acc = __builtin_amdgcn_wmma_f32_16x16x32_f16(false, a, false, b, (short)0, acc, false, false);
}

// gates(16 x 512) = bias + A1*W1^T + A2*W2^T ; this wave covers n-tiles [ntLo, ntLo+NTW)
// NOTE: nt loop pinned to unroll 2 — full unroll spills B-fragments to scratch (round-2 lesson).
template<int NCA, int NCB>
__device__ __forceinline__ void gates_gemm(int lane, int ntLo, float* gbuf,
    const float* __restrict__ bias,
    const v16h* fa, const _Float16* __restrict__ wa,
    const v16h* fb, const _Float16* __restrict__ wb)
{
  const int nl = lane & 15, rb = (lane >> 4) * 8;
  #pragma unroll 2
  for (int i = 0; i < NTW; ++i){
    const int nt = ntLo + i;
    float bv = bias[nt*16 + nl];
    v8f acc;
    #pragma unroll
    for (int r=0;r<8;++r) acc[r] = bv;
    #pragma unroll
    for (int kc=0;kc<NCA;++kc) wmma_acc(acc, fa[kc], wa + (size_t)(kc*32+nt)*512, lane);
    #pragma unroll
    for (int kc=0;kc<NCB;++kc) wmma_acc(acc, fb[kc], wb + (size_t)(kc*32+nt)*512, lane);
    float* gp = gbuf + nt*16 + nl;           // C/D: VGPR r -> row rb+r, col nt*16+nl
    #pragma unroll
    for (int r=0;r<8;++r) gp[(size_t)(rb+r)*GSTR] = acc[r];
  }
}

// LSTM pointwise over 16x128 state, 128 threads (+ optional rank-1 x-term for decoder L0)
// unroll pinned to 4: keeps the 512-iteration recurrent loop body inside the I-cache.
__device__ __forceinline__ void cell_activate(int tid, const float* __restrict__ g,
    float* cbuf, _Float16* hbuf, const float* curv, const float* wih)
{
  #pragma unroll 4
  for (int u=0; u<(BM*H_)/128; ++u){
    int e = tid + (u<<7);
    int row = e >> 7, j = e & 127;           // e == row*128 + j
    const float* gr = g + (size_t)row*GSTR;
    float gi=gr[j], gf=gr[H_+j], gg=gr[2*H_+j], go=gr[3*H_+j];
    if (wih){
      float cu = curv[row];
      gi += cu*wih[j]; gf += cu*wih[H_+j]; gg += cu*wih[2*H_+j]; go += cu*wih[3*H_+j];
    }
    float c = sigm(gf)*cbuf[e] + sigm(gi)*tanh_fast(gg);
    hbuf[row*HSTR + j] = (_Float16)(sigm(go)*tanh_fast(c));
    cbuf[e] = c;
  }
}

// fp32 (512 x Kdim) weight -> f16 tiles in exact WMMA B-fragment order.
// tile = kc*32+nt ; element e = lane*16+half ; value = W[nt*16 + lane%16][kc*32 + (half/8)*16 + (lane/16)*8 + half%8]
__global__ void convert_w(const float* __restrict__ W, _Float16* __restrict__ dst, int Kdim){
  const int tile = blockIdx.x;
  const int kc = tile >> 5, nt = tile & 31;
  for (int e = threadIdx.x; e < 512; e += blockDim.x){
    int ln = e >> 4, hf = e & 15;
    int n = nt*16 + (ln & 15);
    int k = kc*32 + ((hf >> 3) << 4) + ((ln >> 4) << 3) + (hf & 7);
    dst[(size_t)tile*512 + e] = (k < Kdim) ? (_Float16)W[(size_t)n*Kdim + k] : (_Float16)0.f;
  }
}

__global__ __launch_bounds__(32*WAVES) void lstm_fused(
    const float* __restrict__ x_in,  const float* __restrict__ dec_in,
    const float* __restrict__ eb0,   const float* __restrict__ eb1,
    const float* __restrict__ dwih0, const float* __restrict__ db0, const float* __restrict__ db1,
    const float* __restrict__ headW, const float* __restrict__ headb,
    const _Float16* __restrict__ wEih0, const _Float16* __restrict__ wEhh0,
    const _Float16* __restrict__ wEih1, const _Float16* __restrict__ wEhh1,
    const _Float16* __restrict__ wDhh0, const _Float16* __restrict__ wDih1,
    const _Float16* __restrict__ wDhh1,
    float* __restrict__ out)
{
  __shared__ _Float16 s_x [BM*XSTR];
  __shared__ _Float16 s_h0[BM*HSTR];
  __shared__ _Float16 s_h1[BM*HSTR];
  __shared__ float    s_c0[BM*H_];
  __shared__ float    s_c1[BM*H_];
  __shared__ float    s_g [BM*GSTR];
  __shared__ float    s_cur[BM];
  __shared__ float    s_wih[NG];
  __shared__ float    s_hw[H_];

  const int tid  = threadIdx.x;
  const int lane = tid & 31;
  const int ntLo = (tid >> 5) * NTW;
  const int b0   = blockIdx.x * BM;

  for (int e=tid; e<BM*H_;   e+=32*WAVES){ s_c0[e]=0.f; s_c1[e]=0.f; }
  for (int e=tid; e<BM*HSTR; e+=32*WAVES){ s_h0[e]=(_Float16)0.f; s_h1[e]=(_Float16)0.f; }
  for (int e=tid; e<BM*XSTR; e+=32*WAVES) s_x[e]=(_Float16)0.f;   // K-padding 21..31 stays zero
  __syncthreads();

  // ---------------- encoder: 512 recurrent steps ----------------
  for (int t=0; t<T_; ++t){
    for (int e=tid; e<BM*F_; e+=32*WAVES){
      int r = e / F_, f = e - r*F_;
      s_x[r*XSTR + f] = (_Float16)x_in[((size_t)(b0+r)*T_ + t)*F_ + f];
    }
    if (t+1 < T_ && tid < 32){   // prefetch next step's input rows (global_prefetch_b8)
      int r = tid >> 1;
      const char* p = (const char*)&x_in[((size_t)(b0+r)*T_ + (t+1))*F_] + (tid & 1)*64;
      __builtin_prefetch(p, 0, 3);
    }
    __syncthreads();

    v16h ax = load_a_frag(s_x, XSTR, 0, lane);
    v16h ah[4];
    #pragma unroll
    for (int kc=0;kc<4;++kc) ah[kc] = load_a_frag(s_h0, HSTR, kc*32, lane);
    gates_gemm<1,4>(lane, ntLo, s_g, eb0, &ax, wEih0, ah, wEhh0);
    __syncthreads();
    cell_activate(tid, s_g, s_c0, s_h0, nullptr, nullptr);
    __syncthreads();

    v16h ai[4], ar[4];
    #pragma unroll
    for (int kc=0;kc<4;++kc){ ai[kc]=load_a_frag(s_h0,HSTR,kc*32,lane); ar[kc]=load_a_frag(s_h1,HSTR,kc*32,lane); }
    gates_gemm<4,4>(lane, ntLo, s_g, eb1, ai, wEih1, ar, wEhh1);
    __syncthreads();
    cell_activate(tid, s_g, s_c1, s_h1, nullptr, nullptr);
    __syncthreads();
  }

  // ---------------- decoder: 15 steps + scalar head ----------------
  if (tid < BM) s_cur[tid] = dec_in[(size_t)(b0+tid)*HOR_];
  for (int e=tid;e<NG;e+=32*WAVES) s_wih[e] = dwih0[e];
  for (int e=tid;e<H_;e+=32*WAVES) s_hw[e]  = headW[e];
  __syncthreads();
  const float hb = headb[0];

  for (int s=0; s<HOR_; ++s){
    v16h ah[4];
    #pragma unroll
    for (int kc=0;kc<4;++kc) ah[kc] = load_a_frag(s_h0, HSTR, kc*32, lane);
    gates_gemm<0,4>(lane, ntLo, s_g, db0, nullptr, nullptr, ah, wDhh0);
    __syncthreads();
    cell_activate(tid, s_g, s_c0, s_h0, s_cur, s_wih);   // rank-1 x-term folded in
    __syncthreads();

    v16h ai[4], ar[4];
    #pragma unroll
    for (int kc=0;kc<4;++kc){ ai[kc]=load_a_frag(s_h0,HSTR,kc*32,lane); ar[kc]=load_a_frag(s_h1,HSTR,kc*32,lane); }
    gates_gemm<4,4>(lane, ntLo, s_g, db1, ai, wDih1, ar, wDhh1);
    __syncthreads();
    cell_activate(tid, s_g, s_c1, s_h1, nullptr, nullptr);
    __syncthreads();

    if (tid < BM){
      float acc = hb;
      #pragma unroll 4
      for (int k=0;k<H_;++k) acc += (float)s_h1[tid*HSTR + k] * s_hw[k];
      out[(size_t)(b0+tid)*HOR_ + s] = acc;
      s_cur[tid] = acc;
    }
    __syncthreads();
  }
}

extern "C" void kernel_launch(void* const* d_in, const int* in_sizes, int n_in,
                              void* d_out, int out_size, void* d_ws, size_t ws_size,
                              hipStream_t stream)
{
  const float* enc_x = (const float*)d_in[0];
  const float* dec_x = (const float*)d_in[1];
  const float* eWih0 = (const float*)d_in[2];
  const float* eWhh0 = (const float*)d_in[3];
  const float* eb0   = (const float*)d_in[4];
  const float* eWih1 = (const float*)d_in[5];
  const float* eWhh1 = (const float*)d_in[6];
  const float* eb1   = (const float*)d_in[7];
  const float* dWih0 = (const float*)d_in[8];
  const float* dWhh0 = (const float*)d_in[9];
  const float* db0   = (const float*)d_in[10];
  const float* dWih1 = (const float*)d_in[11];
  const float* dWhh1 = (const float*)d_in[12];
  const float* db1   = (const float*)d_in[13];
  const float* headW = (const float*)d_in[14];
  const float* headb = (const float*)d_in[15];

  // f16 swizzled weights in workspace (total 800 KB, L2-resident)
  _Float16* ws    = (_Float16*)d_ws;
  _Float16* wEih0 = ws;                  // 32  tiles * 512
  _Float16* wEhh0 = wEih0 + 32*512;      // 128 tiles * 512
  _Float16* wEih1 = wEhh0 + 128*512;
  _Float16* wEhh1 = wEih1 + 128*512;
  _Float16* wDhh0 = wEhh1 + 128*512;
  _Float16* wDih1 = wDhh0 + 128*512;
  _Float16* wDhh1 = wDih1 + 128*512;

  convert_w<<<32,  256, 0, stream>>>(eWih0, wEih0, F_);
  convert_w<<<128, 256, 0, stream>>>(eWhh0, wEhh0, H_);
  convert_w<<<128, 256, 0, stream>>>(eWih1, wEih1, H_);
  convert_w<<<128, 256, 0, stream>>>(eWhh1, wEhh1, H_);
  convert_w<<<128, 256, 0, stream>>>(dWhh0, wDhh0, H_);
  convert_w<<<128, 256, 0, stream>>>(dWih1, wDih1, H_);
  convert_w<<<128, 256, 0, stream>>>(dWhh1, wDhh1, H_);

  lstm_fused<<<B_/BM, 32*WAVES, 0, stream>>>(enc_x, dec_x, eb0, eb1, dWih0, db0, db1, headW, headb,
      wEih0, wEhh0, wEih1, wEhh1, wDhh0, wDih1, wDhh1, (float*)d_out);
}
